// Attention_27977416966318
// MI455X (gfx1250) — matrix operand order repaired and checked by reading, not compiled, not run on hardware
//
#include <hip/hip_runtime.h>
#include <math.h>
#include <stdint.h>

typedef __bf16 bf16;
typedef __attribute__((ext_vector_type(16))) __bf16 v16bf;
typedef __attribute__((ext_vector_type(8)))  float   v8f;
typedef __attribute__((ext_vector_type(4)))  float   v4f;
typedef __attribute__((ext_vector_type(4)))  int     v4i;

#define B_   64
#define S_   2048
#define H_   256
#define H3_  768
#define KSD  512            // K for static+dynamic part of the GEMM
#define MT   128            // rows (positions) per workgroup
#define NT   16             // n-tile width
#define NTILES (H3_/NT)     // 48
#define KSTEPS (KSD/32)     // 16 wmma k-steps

// --------------------------------------------------------------------------
// CDNA5 async global->LDS copy path (direct, ASYNCcnt-tracked, no VGPR data)
// with a safe fallback to a VGPR-mediated copy if the builtin is absent.
// Probe result: param 0 is an AS1 (global) pointer to 'int __vector(4)'.
// --------------------------------------------------------------------------
#if __has_builtin(__builtin_amdgcn_global_load_async_to_lds_b128)
#define USE_ASYNC_LDS 1
typedef __attribute__((address_space(1))) v4i gas_v4i;
typedef __attribute__((address_space(3))) v4i las_v4i;
#endif

__device__ __forceinline__ void copy16_g2l(const void* gsrc, void* ldst) {
#if defined(USE_ASYNC_LDS)
    __builtin_amdgcn_global_load_async_to_lds_b128(
        (gas_v4i*)(uintptr_t)gsrc,
        (las_v4i*)(uint32_t)(uintptr_t)ldst,
        0, 0);
#else
    *(uint4*)ldst = *(const uint4*)gsrc;
#endif
}

__device__ __forceinline__ void wait_async0() {
#if __has_builtin(__builtin_amdgcn_s_wait_asynccnt)
    __builtin_amdgcn_s_wait_asynccnt(0);
#else
    asm volatile("s_wait_asynccnt 0x0" ::: "memory");
#endif
}

// ---------------------------------------------------------------------------
// Repack W[:, 0:512] transposed to bf16, WMMA-B-fragment friendly:
// element (k, o) -> Wp[((k>>4)*768 + o)*16 + (k&15)]
// so a lane's 16 contiguous K values for a fixed o are one 32B chunk.
// ---------------------------------------------------------------------------
__global__ __launch_bounds__(256) void prep_w(const float* __restrict__ W,
                                              bf16* __restrict__ Wp) {
    int idx = blockIdx.x * 256 + threadIdx.x;   // over 512*768 elements
    if (idx >= KSD * H3_) return;
    int o = idx % H3_;
    int k = idx / H3_;
    float w = W[o * H3_ + k];
    Wp[((size_t)(k >> 4) * H3_ + o) * 16 + (k & 15)] = (bf16)w;
}

// ---------------------------------------------------------------------------
// bias[b][o] = sum_k W[o][512+k] * dec[b][k]   (f32, keeps 1/3 of K exact)
// ---------------------------------------------------------------------------
__global__ __launch_bounds__(256) void prep_bias(const float* __restrict__ W,
                                                 const float* __restrict__ dec,
                                                 float* __restrict__ bias) {
    __shared__ float ds[H_];
    const int b = blockIdx.x;
    const int o = blockIdx.y * 256 + threadIdx.x;
    ds[threadIdx.x] = dec[b * H_ + threadIdx.x];
    __syncthreads();
    const float* wr = W + (size_t)o * H3_ + KSD;
    float s = 0.f;
    #pragma unroll 8
    for (int k = 0; k < H_; ++k) s += wr[k] * ds[k];
    bias[b * H3_ + o] = s;
}

// ---------------------------------------------------------------------------
// Stage one 16KB B n-tile (32 k-groups x 512B) into an LDS buffer.
// 256 threads x 4 x 16B chunks, all via the async direct-to-LDS path.
// ---------------------------------------------------------------------------
__device__ __forceinline__ void stage_B(const bf16* __restrict__ Wp,
                                        bf16* __restrict__ Bsbuf,
                                        int n0, int tid) {
    #pragma unroll
    for (int j = 0; j < 4; ++j) {
        const int f  = tid + j * 256;   // 16B-chunk index 0..1023
        const int g  = f >> 5;          // k-group 0..31
        const int wi = f & 31;          // 16B chunk within group
        const uint4* src = (const uint4*)(Wp + ((size_t)g * H3_ + n0) * 16) + wi;
        copy16_g2l(src, (uint4*)Bsbuf + f);
    }
}

// ---------------------------------------------------------------------------
// Main fused kernel: pre-softmax scores[b*s] = sum_o v[o]*tanh(GEMM + bias)
// 128 rows per WG, 8 waves x 16 rows. A strip lives in registers (read HBM
// exactly once); B n-tiles double-buffered in LDS via async copies so the
// 16KB stage fully overlaps the 16-WMMA + tanh compute of the previous tile.
// ---------------------------------------------------------------------------
__global__ __launch_bounds__(256) void attn_gemm(
    const float* __restrict__ stat, const float* __restrict__ dyn,
    const bf16* __restrict__ Wp,    const float* __restrict__ bias,
    const float* __restrict__ v,    float* __restrict__ scores)
{
    __shared__ bf16 Bs[2][KSD * NT];       // 2 x 16 KB double buffer

    const int tid  = threadIdx.x;
    const int wave = tid >> 5;
    const int lane = tid & 31;
    const int li   = lane & 15;            // N-lane / A-row selector
    const int hh   = lane >> 4;            // half selector

    const int rowBase = blockIdx.x * MT;   // global position base of tile
    const int b       = rowBase / S_;      // MT divides S_, so single b per WG
    const int myRow   = rowBase + wave * 16 + li;

    // prefetch first B tile before the (long-latency) A register fill
    stage_B(Wp, Bs[0], 0, tid);

    // ---- load this wave's A strip (16 rows x 512 K) to registers as bf16 ----
    v16bf areg[KSTEPS];
    {
        const float* srow_s = stat + (size_t)myRow * H_;
        const float* srow_d = dyn  + (size_t)myRow * H_;
        #pragma unroll
        for (int kk = 0; kk < KSTEPS; ++kk) {
            const int k0  = kk * 32;
            const int klo = k0 + hh * 16;    // lanes 0-15: k0   ; 16-31: k0+16
            const int khi = klo + 8;         // lanes 0-15: k0+8 ; 16-31: k0+24
            const float* p0 = (klo < H_) ? (srow_s + klo) : (srow_d + klo - H_);
            const float* p1 = (khi < H_) ? (srow_s + khi) : (srow_d + khi - H_);
            v4f a0 = *(const v4f*)(p0);
            v4f a1 = *(const v4f*)(p0 + 4);
            v4f a2 = *(const v4f*)(p1);
            v4f a3 = *(const v4f*)(p1 + 4);
            v16bf a;
            #pragma unroll
            for (int i = 0; i < 4; ++i) {
                a[i]      = (bf16)a0[i];
                a[4 + i]  = (bf16)a1[i];
                a[8 + i]  = (bf16)a2[i];
                a[12 + i] = (bf16)a3[i];
            }
            areg[kk] = a;
        }
    }

    float sp[8] = {0.f, 0.f, 0.f, 0.f, 0.f, 0.f, 0.f, 0.f};

    for (int nt = 0; nt < NTILES; ++nt) {
        const int buf = nt & 1;

#if defined(USE_ASYNC_LDS)
        wait_async0();                      // my async writes to Bs[buf] done
#endif
        __syncthreads();                    // everyone's writes done; everyone
                                            // finished reading Bs[buf^1]

        if (nt + 1 < NTILES)                // prefetch next tile into other buf
            stage_B(Wp, Bs[buf ^ 1], (nt + 1) * NT, tid);

        const int n0 = nt * NT;
        const float vv = v[n0 + li];
        const float bb = bias[b * H3_ + n0 + li];
        v8f acc = {bb, bb, bb, bb, bb, bb, bb, bb};

        #pragma unroll
        for (int kk = 0; kk < KSTEPS; ++kk) {
            const int gl = kk * 2 + hh;     // lane's 16 contiguous K values
            const v16bf bfrag = *(const v16bf*)(Bs[buf] + ((size_t)gl * NT + li) * 16);
            acc = __builtin_amdgcn_wmma_f32_16x16x32_bf16(
                      false, areg[kk], false, bfrag,
                      (short)0, acc, false, false);
        }

        // fused epilogue: tanh + v-weighted partial reduction over this n-tile
        #pragma unroll
        for (int r = 0; r < 8; ++r) sp[r] += vv * tanhf(acc[r]);
    }

    // reduce over the 16 lanes of each half (masks < 16 never cross halves)
    #pragma unroll
    for (int off = 1; off < 16; off <<= 1) {
        #pragma unroll
        for (int r = 0; r < 8; ++r) sp[r] += __shfl_xor(sp[r], off, 32);
    }

    // C/D layout: VGPR r, lanes 0-15 -> row r ; lanes 16-31 -> row r+8
    if (li < 8) scores[rowBase + wave * 16 + hh * 8 + li] = sp[li];
}

// ---------------------------------------------------------------------------
// Row softmax over S=2048, one block per b, in-place on d_out.
// ---------------------------------------------------------------------------
__global__ __launch_bounds__(256) void softmax_rows(float* __restrict__ scores) {
    __shared__ float red[256];
    const int b   = blockIdx.x;
    const int tid = threadIdx.x;
    float* row = scores + (size_t)b * S_;

    float x[8];
    float m = -3.402823466e38f;
    #pragma unroll
    for (int j = 0; j < 8; ++j) { x[j] = row[tid + j * 256]; m = fmaxf(m, x[j]); }
    red[tid] = m; __syncthreads();
    for (int s = 128; s > 0; s >>= 1) {
        if (tid < s) red[tid] = fmaxf(red[tid], red[tid + s]);
        __syncthreads();
    }
    m = red[0]; __syncthreads();

    float sum = 0.f;
    #pragma unroll
    for (int j = 0; j < 8; ++j) { x[j] = __expf(x[j] - m); sum += x[j]; }
    red[tid] = sum; __syncthreads();
    for (int s = 128; s > 0; s >>= 1) {
        if (tid < s) red[tid] += red[tid + s];
        __syncthreads();
    }
    const float inv = 1.f / red[0];
    #pragma unroll
    for (int j = 0; j < 8; ++j) row[tid + j * 256] = x[j] * inv;
}

// ---------------------------------------------------------------------------
extern "C" void kernel_launch(void* const* d_in, const int* in_sizes, int n_in,
                              void* d_out, int out_size, void* d_ws, size_t ws_size,
                              hipStream_t stream) {
    (void)in_sizes; (void)n_in; (void)out_size; (void)ws_size;

    const float* stat = (const float*)d_in[0];   // [B,S,H]
    const float* dyn  = (const float*)d_in[1];   // [B,S,H]
    const float* dec  = (const float*)d_in[2];   // [B,H]
    const float* v    = (const float*)d_in[3];   // [1,H3]
    const float* W    = (const float*)d_in[4];   // [H3,H3]
    float* out = (float*)d_out;                  // [B,S]

    bf16*  Wp   = (bf16*)d_ws;                                   // 512*768 bf16
    float* bias = (float*)((char*)d_ws + (size_t)KSD * H3_ * sizeof(bf16));

    prep_w   <<<dim3((KSD * H3_ + 255) / 256), 256, 0, stream>>>(W, Wp);
    prep_bias<<<dim3(B_, H3_ / 256),           256, 0, stream>>>(W, dec, bias);
    attn_gemm<<<dim3((B_ * S_) / MT),          256, 0, stream>>>(stat, dyn, Wp,
                                                                 bias, v, out);
    softmax_rows<<<dim3(B_),                   256, 0, stream>>>(out);
}
